// NTM_72524817760563
// MI455X (gfx1250) — matrix operand order repaired and checked by reading, not compiled, not run on hardware
//
#include <hip/hip_runtime.h>

#define B_  1024
#define T_  64
#define IN_ 8
#define H_  128
#define M_  128
#define D_  20
#define BT  16
#define NWG (B_ / BT)       // 64 workgroups, 16 batches each
#define NTHREADS 256        // 8 waves of 32

typedef __attribute__((ext_vector_type(16))) _Float16 v16h;
typedef __attribute__((ext_vector_type(8)))  float    v8f;
typedef _Float16 h16;

// ---------------- WMMA fragment loaders (CDNA5 wave32 layouts) ----------------

// A matrix 16x32 f16: lanes 0-15 = rows, K pattern {kb+off..kb+off+7, kb+16+off..kb+16+off+7}
__device__ __forceinline__ v16h load_fragA(const h16* p, int ld, int r, int kbase,
                                           int kvalid, int lane) {
  const int off = (lane >> 4) << 3;
  const h16* q = p + r * ld + kbase + off;
  v16h v;
#pragma unroll
  for (int i = 0; i < 8; ++i) {
    int k0 = kbase + off + i;
    int k1 = k0 + 16;
    v[i]     = (k0 < kvalid) ? q[i]      : (h16)0.0f;
    v[i + 8] = (k1 < kvalid) ? q[i + 16] : (h16)0.0f;
  }
  return v;
}

// B matrix 32x16 f16: lane holds column n = lane&15; K pattern {kb+off..kb+off+7, kb+16+off..kb+16+off+7}, off = 8*(lane>>4)
__device__ __forceinline__ v16h load_fragB(const h16* p, int ld, int n, int kbase,
                                           int kvalid, int lane) {
  const int koff = kbase + ((lane >> 4) << 3);
  const h16* q = p + n * ld + koff;
  v16h v;
#pragma unroll
  for (int i = 0; i < 8; ++i) {
    v[i]     = ((koff + i) < kvalid) ? q[i] : (h16)0.0f;
    v[i + 8] = ((koff + i + 16) < kvalid) ? q[i + 16] : (h16)0.0f;
  }
  return v;
}

__device__ __forceinline__ void wmma_acc(v8f& c, const h16* A, int lda, int Kloop,
                                         int kvalid, const h16* W, int ldw, int nc,
                                         int lane) {
  for (int kb = 0; kb < Kloop; kb += 32) {
    v16h a = load_fragA(A, lda, lane & 15, kb, kvalid, lane);
    v16h b = load_fragB(W, ldw, nc, kb, kvalid, lane);
    c = __builtin_amdgcn_wmma_f32_16x16x32_f16(false, a, false, b, (short)0, c,
                                               false, false);
  }
}

__device__ __forceinline__ v8f tile_bias(const float* b0, const float* b1, int nc) {
  float bi = 0.f;
  if (b0) bi += b0[nc];
  if (b1) bi += b1[nc];
  v8f c;
#pragma unroll
  for (int i = 0; i < 8; ++i) c[i] = bi;
  return c;
}

// C/D layout: lane l -> col n = l&15, rows m = r + 8*(l>>4)
__device__ __forceinline__ void store_tile_f32(const v8f& c, float* out, int ldo,
                                               int n, int nmax, int lane) {
  if (n < nmax) {
    int rb = (lane >> 4) << 3;
#pragma unroll
    for (int r = 0; r < 8; ++r) out[(r + rb) * ldo + n] = c[r];
  }
}

__device__ __forceinline__ void store_tile_h16(const v8f& c, h16* out, int ldo,
                                               int n, int nmax, int lane) {
  if (n < nmax) {
    int rb = (lane >> 4) << 3;
#pragma unroll
    for (int r = 0; r < 8; ++r) out[(r + rb) * ldo + n] = (h16)c[r];
  }
}

// row-wise softmax over 16 x 128 LDS tile; red is 16x16 scratch
__device__ __forceinline__ void row_softmax(float* m, float* red, int tid) {
  int b = tid >> 4, s = tid & 15;
  float mx = -1e30f;
  for (int i = 0; i < 8; ++i) mx = fmaxf(mx, m[b * M_ + s * 8 + i]);
  red[b * 16 + s] = mx;
  __syncthreads();
  if (s == 0) {
    float v = red[b * 16];
    for (int i = 1; i < 16; ++i) v = fmaxf(v, red[b * 16 + i]);
    red[b * 16] = v;
  }
  __syncthreads();
  float rm = red[b * 16];
  float sum = 0.f;
  for (int i = 0; i < 8; ++i) {
    float e = __expf(m[b * M_ + s * 8 + i] - rm);
    m[b * M_ + s * 8 + i] = e;
    sum += e;
  }
  __syncthreads();
  red[b * 16 + s] = sum;
  __syncthreads();
  if (s == 0) {
    float v = 0.f;
    for (int i = 0; i < 16; ++i) v += red[b * 16 + i];
    red[b * 16] = v;
  }
  __syncthreads();
  float inv = 1.f / red[b * 16];
  for (int i = 0; i < 8; ++i) m[b * M_ + s * 8 + i] *= inv;
  __syncthreads();
}

struct HeadW {
  const h16* kw; const float* kb;
  const float* bw; const float* bb;
  const float* gw; const float* gb;
  const h16* sw; const float* sb;
};

// full NTM addressing head; wS (16x128 LDS) is prev weights in, new weights out
__device__ __forceinline__ void addressing(const HeadW& P, const h16* hH,
                                           const h16* mlH, h16* keyH, float* bufA,
                                           float* bufB, float* bufC, h16* bufD,
                                           float* betaS, float* gS, float* red,
                                           float* wS, int tid, int wave, int lane) {
  // key = relu(h Wk^T + bk) -> f16 16x32 (cols >= D zeroed)
  if (wave < 2) {
    int n = wave * 16 + (lane & 15);
    int nc = (n < D_) ? n : 0;
    v8f c = tile_bias(P.kb, nullptr, nc);
    wmma_acc(c, hH, H_, H_, H_, P.kw, H_, nc, lane);
    int rb = (lane >> 4) << 3;
#pragma unroll
    for (int r = 0; r < 8; ++r) {
      float v = (n < D_) ? fmaxf(c[r], 0.f) : 0.f;
      keyH[(r + rb) * 32 + n] = (h16)v;
    }
  }
  __syncthreads();
  // beta / g per-batch scalars (VALU dot over H)
  if (tid < 32) {
    int b = tid & 15;
    const float* w = (tid < 16) ? P.bw : P.gw;
    float acc = (tid < 16) ? P.bb[0] : P.gb[0];
    for (int k = 0; k < H_; ++k) acc += (float)hH[b * H_ + k] * w[k];
    if (tid < 16) betaS[b] = acc; else gS[b] = acc;
  }
  __syncthreads();
  // cos = key @ memLast^T  (B operand = memLast (M x D) shared snapshot)
  {
    int n = wave * 16 + (lane & 15);
    v8f c = {};
    wmma_acc(c, keyH, 32, 32, D_, mlH, D_, n, lane);
    store_tile_f32(c, bufA, M_, n, M_, lane);
  }
  __syncthreads();
  {
    int b = tid >> 4, s = tid & 15;
    float be = betaS[b];
    for (int i = 0; i < 8; ++i) bufA[b * M_ + s * 8 + i] *= be;
  }
  __syncthreads();
  row_softmax(bufA, red, tid);  // bufA = w_content
  // w_g = g*wc + (1-g)*prev
  {
    int b = tid >> 4, s = tid & 15;
    float gg = gS[b];
    for (int i = 0; i < 8; ++i) {
      int j = s * 8 + i;
      bufB[b * M_ + j] = gg * bufA[b * M_ + j] + (1.f - gg) * wS[b * M_ + j];
    }
  }
  __syncthreads();
  // shift = softmax(h Ws^T + bs)
  {
    int n = wave * 16 + (lane & 15);
    v8f c = tile_bias(P.sb, nullptr, n);
    wmma_acc(c, hH, H_, H_, H_, P.sw, H_, n, lane);
    store_tile_f32(c, bufC, M_, n, M_, lane);
  }
  __syncthreads();
  row_softmax(bufC, red, tid);
  // circular conv: circ[b][i] = sum_j shift[b][(i-j)&127] * wg[b][j]
  {
    int b = tid >> 4, s = tid & 15;
    for (int ii = 0; ii < 8; ++ii) {
      int i = s * 8 + ii;
      float acc = 0.f;
      for (int j = 0; j < M_; ++j)
        acc += bufC[b * M_ + ((i - j) & (M_ - 1))] * bufB[b * M_ + j];
      bufD[b * M_ + i] = (h16)acc;
    }
  }
  __syncthreads();
  // sharpen: pow(gamma = g, faithful bug) + normalize -> wS
  {
    int b = tid >> 4, s = tid & 15;
    float gg = gS[b];
    float sum = 0.f;
    for (int i = 0; i < 8; ++i) {
      int j = s * 8 + i;
      float v = powf((float)bufD[b * M_ + j], gg);
      bufA[b * M_ + j] = v;
      sum += v;
    }
    red[b * 16 + s] = sum;
  }
  __syncthreads();
  if ((tid & 15) == 0) {
    int b = tid >> 4;
    float v = 0.f;
    for (int i = 0; i < 16; ++i) v += red[b * 16 + i];
    red[b * 16] = v;
  }
  __syncthreads();
  {
    int b = tid >> 4, s = tid & 15;
    float inv = 1.f / red[b * 16];
    for (int i = 0; i < 8; ++i) {
      int j = s * 8 + i;
      wS[b * M_ + j] = bufA[b * M_ + j] * inv;
    }
  }
  __syncthreads();
}

// -------------------------------- step kernel --------------------------------

struct StepArgs {
  float *hG, *wrG, *wwG, *memG, *mlCur, *mlNext, *lossAcc;
  const float *x, *y;
  HeadW rd, wr;
  const h16 *ew, *aw; const float *eb, *ab;
  const h16 *wxu, *whu, *wxr, *whr, *wxh, *whh;
  const float *bxu, *bhu, *bxr, *bhr, *bxh, *bhh;
  const h16 *rsec, *ssec; const float *rsecb, *ssecb;
  const h16 *outw; const float *outb;
  int t;
};

__global__ void __launch_bounds__(NTHREADS, 1) ntm_step(StepArgs A) {
  __shared__ h16   hH[BT * H_];
  __shared__ h16   xH[BT * 32];
  __shared__ h16   mlH[M_ * D_];
  __shared__ h16   keyH[BT * 32];
  __shared__ h16   readH[BT * 32];
  __shared__ float bufA[BT * M_], bufB[BT * M_], bufC[BT * M_];
  __shared__ h16   bufD[BT * M_];
  __shared__ float wrS[BT * M_], wwS[BT * M_];
  __shared__ float eraseS[BT * 32], addS[BT * 32];
  __shared__ float betaS[BT], gS[BT];
  __shared__ float red[BT * 16];
  __shared__ float lossRed[32];

  const int tid = threadIdx.x;
  const int wave = tid >> 5, lane = tid & 31;
  const int wg = blockIdx.x;
  const int bb = wg * BT;
  const int t = A.t;

  for (int e = tid; e < BT * H_; e += NTHREADS) {   // H_ == M_
    hH[e]  = (h16)A.hG[bb * H_ + e];
    wrS[e] = A.wrG[bb * M_ + e];
    wwS[e] = A.wwG[bb * M_ + e];
  }
  for (int e = tid; e < M_ * D_; e += NTHREADS) mlH[e] = (h16)A.mlCur[e];
  for (int e = tid; e < BT * 32; e += NTHREADS) {
    int b = e >> 5, i = e & 31;
    xH[e] = (i < IN_) ? (h16)A.x[((bb + b) * T_ + t) * IN_ + i] : (h16)0.0f;
    readH[e] = (h16)0.0f;
  }
  __syncthreads();

  // ---- GRU cell ----
  {
    int n = wave * 16 + (lane & 15);
    v8f c = tile_bias(A.bxu, A.bhu, n);                  // z pre-act
    wmma_acc(c, xH, 32, 32, IN_, A.wxu, IN_, n, lane);
    wmma_acc(c, hH, H_, H_, H_, A.whu, H_, n, lane);
    store_tile_f32(c, bufA, H_, n, H_, lane);
    c = tile_bias(A.bxr, A.bhr, n);                      // r pre-act
    wmma_acc(c, xH, 32, 32, IN_, A.wxr, IN_, n, lane);
    wmma_acc(c, hH, H_, H_, H_, A.whr, H_, n, lane);
    store_tile_f32(c, bufB, H_, n, H_, lane);
    c = tile_bias(A.bxh, nullptr, n);                    // x-part
    wmma_acc(c, xH, 32, 32, IN_, A.wxh, IN_, n, lane);
    store_tile_f32(c, bufC, H_, n, H_, lane);
    c = tile_bias(A.bhh, nullptr, n);                    // h-part
    wmma_acc(c, hH, H_, H_, H_, A.whh, H_, n, lane);
    store_tile_h16(c, bufD, H_, n, H_, lane);
  }
  __syncthreads();
  for (int e = tid; e < BT * H_; e += NTHREADS) {
    float z = 1.f / (1.f + __expf(-bufA[e]));
    float r = 1.f / (1.f + __expf(-bufB[e]));
    float hv = (1.f - z) * (float)hH[e] + z * tanhf(bufC[e] + (float)bufD[e] * r);
    bufA[e] = hv;  // stage new h, then commit after barrier
  }
  __syncthreads();
  for (int e = tid; e < BT * H_; e += NTHREADS) hH[e] = (h16)bufA[e];
  __syncthreads();

  // ---- read head addressing, then read (pre-write memory) ----
  addressing(A.rd, hH, mlH, keyH, bufA, bufB, bufC, bufD, betaS, gS, red, wrS,
             tid, wave, lane);
  for (int e = tid; e < BT * D_; e += NTHREADS) {
    int b = e / D_, d = e - b * D_;
    const float* mp = A.memG + ((bb + b) * M_) * D_ + d;
    float acc = 0.f;
    for (int m = 0; m < M_; ++m) acc += wrS[b * M_ + m] * mp[m * D_];
    readH[b * 32 + d] = (h16)acc;
  }
  __syncthreads();

  // ---- write head addressing ----
  addressing(A.wr, hH, mlH, keyH, bufA, bufB, bufC, bufD, betaS, gS, red, wwS,
             tid, wave, lane);

  // ---- erase / add vectors ----
  if (wave < 4) {
    int which = wave >> 1;  // 0 = erase, 1 = add
    int n = (wave & 1) * 16 + (lane & 15);
    int nc = (n < D_) ? n : 0;
    v8f c = tile_bias(which ? A.ab : A.eb, nullptr, nc);
    wmma_acc(c, hH, H_, H_, H_, which ? A.aw : A.ew, H_, nc, lane);
    store_tile_f32(c, which ? addS : eraseS, 32, n, D_, lane);
  }
  __syncthreads();

  // ---- memory update (and memLast snapshot for next step) ----
  const bool lastWG = (wg == NWG - 1);
  for (int e = tid; e < BT * M_ * D_; e += NTHREADS) {
    int b = e / (M_ * D_);
    int rem = e - b * (M_ * D_);
    int m = rem / D_, d = rem - m * D_;
    int gi = ((bb + b) * M_ + m) * D_ + d;
    float w = wwS[b * M_ + m];
    float v = A.memG[gi];
    v = v * (1.f - w * eraseS[b * 32 + d]) + w * addS[b * 32 + d];
    A.memG[gi] = v;
    if (lastWG && b == BT - 1) A.mlNext[rem] = v;
  }
  __syncthreads();

  // ---- h = tanh(read_sec(read) + self_sec(h)) ----
  {
    int n = wave * 16 + (lane & 15);
    v8f c = tile_bias(A.rsecb, A.ssecb, n);
    wmma_acc(c, readH, 32, 32, D_, A.rsec, D_, n, lane);
    wmma_acc(c, hH, H_, H_, H_, A.ssec, H_, n, lane);
    store_tile_f32(c, bufA, H_, n, H_, lane);
  }
  __syncthreads();
  for (int e = tid; e < BT * H_; e += NTHREADS) hH[e] = (h16)tanhf(bufA[e]);
  __syncthreads();

  // ---- out = out(h); loss += sum((out - y)^2) ----
  if (tid < 32) lossRed[tid] = 0.f;
  __syncthreads();
  if (wave == 0) {
    int n = lane & 15;
    int nc = (n < IN_) ? n : 0;
    v8f c = tile_bias(A.outb, nullptr, nc);
    wmma_acc(c, hH, H_, H_, H_, A.outw, H_, nc, lane);
    float part = 0.f;
    if (n < IN_) {
      int rb = (lane >> 4) << 3;
#pragma unroll
      for (int r = 0; r < 8; ++r) {
        int m = r + rb;
        float d = c[r] - A.y[((bb + m) * T_ + t) * IN_ + n];
        part += d * d;
      }
    }
    lossRed[lane] = part;
  }
  __syncthreads();
  if (tid == 0) {
    float s = 0.f;
    for (int i = 0; i < 32; ++i) s += lossRed[i];
    A.lossAcc[wg] += s;  // per-WG accumulator: deterministic across launches
  }

  // ---- commit recurrent state ----
  for (int e = tid; e < BT * H_; e += NTHREADS) {
    A.hG[bb * H_ + e]  = (float)hH[e];
    A.wrG[bb * M_ + e] = wrS[e];
    A.wwG[bb * M_ + e] = wwS[e];
  }
}

// ------------------------- prep (init + f16 weight packs) -------------------------

struct PrepArgs {
  const float *kwr, *swr, *kww, *sww, *ew, *aw;
  const float *wxu, *whu, *wxr, *whr, *wxh, *whh, *rsec, *ssec, *outw;
  h16 *kwr16, *swr16, *kww16, *sww16, *ew16, *aw16;
  h16 *wxu16, *whu16, *wxr16, *whr16, *wxh16, *whh16, *rsec16, *ssec16, *outw16;
  const float *h_init, *rw_init, *ww_init;
  float *hG, *wrG, *wwG, *memG, *ml0, *ml1, *lossAcc;
};

__global__ void ntm_prep(PrepArgs P) {
  const int idx = blockIdx.x * blockDim.x + threadIdx.x;
  const int stride = gridDim.x * blockDim.x;
#define CVT(S, D, N) for (int e = idx; e < (N); e += stride) D[e] = (h16)S[e];
  CVT(P.kwr, P.kwr16, D_ * H_)
  CVT(P.swr, P.swr16, M_ * H_)
  CVT(P.kww, P.kww16, D_ * H_)
  CVT(P.sww, P.sww16, M_ * H_)
  CVT(P.ew,  P.ew16,  D_ * H_)
  CVT(P.aw,  P.aw16,  D_ * H_)
  CVT(P.wxu, P.wxu16, H_ * IN_)
  CVT(P.whu, P.whu16, H_ * H_)
  CVT(P.wxr, P.wxr16, H_ * IN_)
  CVT(P.whr, P.whr16, H_ * H_)
  CVT(P.wxh, P.wxh16, H_ * IN_)
  CVT(P.whh, P.whh16, H_ * H_)
  CVT(P.rsec, P.rsec16, H_ * D_)
  CVT(P.ssec, P.ssec16, H_ * H_)
  CVT(P.outw, P.outw16, IN_ * H_)
#undef CVT
  for (int e = idx; e < B_ * H_; e += stride) P.hG[e] = P.h_init[e & (H_ - 1)];
  for (int e = idx; e < B_ * M_; e += stride) {
    P.wrG[e] = P.rw_init[e & (M_ - 1)];
    P.wwG[e] = P.ww_init[e & (M_ - 1)];
  }
  for (int e = idx; e < B_ * M_ * D_; e += stride) P.memG[e] = 0.f;
  for (int e = idx; e < M_ * D_; e += stride) { P.ml0[e] = 0.f; P.ml1[e] = 0.f; }
  for (int e = idx; e < NWG; e += stride) P.lossAcc[e] = 0.f;
}

__global__ void ntm_finish(const float* lossAcc, float* out) {
  if (threadIdx.x == 0 && blockIdx.x == 0) {
    float s = 0.f;
    for (int i = 0; i < NWG; ++i) s += lossAcc[i];
    out[0] = s;
  }
}

// -------------------------------- host launcher --------------------------------

extern "C" void kernel_launch(void* const* d_in, const int* in_sizes, int n_in,
                              void* d_out, int out_size, void* d_ws, size_t ws_size,
                              hipStream_t stream) {
  (void)in_sizes; (void)n_in; (void)out_size; (void)ws_size;
  auto F = [&](int i) { return (const float*)d_in[i]; };

  // workspace carve-up (floats, then f16 pool)
  float* ws = (float*)d_ws;
  float* hG   = ws;                       // B*H
  float* wrG  = hG  + B_ * H_;            // B*M
  float* wwG  = wrG + B_ * M_;            // B*M
  float* memG = wwG + B_ * M_;            // B*M*D
  float* ml0  = memG + B_ * M_ * D_;      // M*D
  float* ml1  = ml0 + M_ * D_;            // M*D
  float* lossAcc = ml1 + M_ * D_;         // NWG
  h16* hp = (h16*)(lossAcc + NWG);
  h16 *kwr16 = hp;            h16 *swr16 = kwr16 + D_ * H_;
  h16 *kww16 = swr16 + M_*H_; h16 *sww16 = kww16 + D_ * H_;
  h16 *ew16  = sww16 + M_*H_; h16 *aw16  = ew16 + D_ * H_;
  h16 *wxu16 = aw16 + D_*H_;  h16 *whu16 = wxu16 + H_ * IN_;
  h16 *wxr16 = whu16 + H_*H_; h16 *whr16 = wxr16 + H_ * IN_;
  h16 *wxh16 = whr16 + H_*H_; h16 *whh16 = wxh16 + H_ * IN_;
  h16 *rsec16 = whh16 + H_*H_; h16 *ssec16 = rsec16 + H_ * D_;
  h16 *outw16 = ssec16 + H_*H_;

  PrepArgs P;
  P.kwr = F(2);  P.swr = F(8);  P.kww = F(12); P.sww = F(18);
  P.ew = F(22);  P.aw = F(24);
  P.wxu = F(26); P.whu = F(28); P.wxr = F(30); P.whr = F(32);
  P.wxh = F(34); P.whh = F(36); P.rsec = F(38); P.ssec = F(40); P.outw = F(42);
  P.kwr16 = kwr16; P.swr16 = swr16; P.kww16 = kww16; P.sww16 = sww16;
  P.ew16 = ew16; P.aw16 = aw16;
  P.wxu16 = wxu16; P.whu16 = whu16; P.wxr16 = wxr16; P.whr16 = whr16;
  P.wxh16 = wxh16; P.whh16 = whh16; P.rsec16 = rsec16; P.ssec16 = ssec16;
  P.outw16 = outw16;
  P.h_init = F(44); P.rw_init = F(45); P.ww_init = F(46);
  P.hG = hG; P.wrG = wrG; P.wwG = wwG; P.memG = memG;
  P.ml0 = ml0; P.ml1 = ml1; P.lossAcc = lossAcc;
  ntm_prep<<<512, 256, 0, stream>>>(P);

  StepArgs S;
  S.hG = hG; S.wrG = wrG; S.wwG = wwG; S.memG = memG; S.lossAcc = lossAcc;
  S.x = F(0); S.y = F(1);
  S.rd = HeadW{kwr16, F(3), F(4), F(5), F(6), F(7), swr16, F(9)};
  S.wr = HeadW{kww16, F(13), F(14), F(15), F(16), F(17), sww16, F(19)};
  S.ew = ew16; S.eb = F(23); S.aw = aw16; S.ab = F(25);
  S.wxu = wxu16; S.bxu = F(27); S.whu = whu16; S.bhu = F(29);
  S.wxr = wxr16; S.bxr = F(31); S.whr = whr16; S.bhr = F(33);
  S.wxh = wxh16; S.bxh = F(35); S.whh = whh16; S.bhh = F(37);
  S.rsec = rsec16; S.rsecb = F(39); S.ssec = ssec16; S.ssecb = F(41);
  S.outw = outw16; S.outb = F(43);

  for (int t = 0; t < T_; ++t) {
    S.t = t;
    S.mlCur = (t & 1) ? ml1 : ml0;   // snapshot of mem[B-1] at step start
    S.mlNext = (t & 1) ? ml0 : ml1;  // written by WG owning batch B-1
    ntm_step<<<NWG, NTHREADS, 0, stream>>>(S);
  }
  ntm_finish<<<1, 32, 0, stream>>>(lossAcc, (float*)d_out);
}
